// EquivariantGraphAttention_42305427866206
// MI455X (gfx1250) — compile-verified, matrix operand order checked by reading, not run on hardware
//
#include <hip/hip_runtime.h>
#include <hip/hip_bf16.h>
#include <stdint.h>

#define EDGES 32768
#define NNODE 8192

typedef __bf16 bf16;
typedef __attribute__((ext_vector_type(16))) __bf16 v16bf;
typedef __attribute__((ext_vector_type(8)))  __bf16 v8bf;
typedef __attribute__((ext_vector_type(8)))  float  v8f;

// row -> source-buffer index (0=w0 path, 1=m1, 2=m2, 3=m3)
__constant__ int d_sb[16]  = {0,1,0,1,2,1,0,1,2,3,2,1,0,1,2,3};
// so2a: element offset of this coefficient row inside its source buffer (128-ch blocks, +384 extra on w0)
__constant__ int d_soA[16] = {384,384,512,0,256,512,640,128,0,128,384,640,768,256,128,0};
// so2b: same, no extra region
__constant__ int d_soB[16] = {0,384,128,0,256,512,256,128,0,128,384,640,384,256,128,0};
__constant__ int d_strideY[4] = {896,768,512,256};
__constant__ int d_strideZ[4] = {512,768,512,256};
__constant__ int d_lof[16] = {0,1,1,1,2,2,2,2,2,3,3,3,3,3,3,3};

// ---------------------------------------------------------------- helpers
__device__ __forceinline__ float sigf(float x) { return 1.f / (1.f + __expf(-x)); }
__device__ __forceinline__ float wsum(float v) {
  #pragma unroll
  for (int i = 16; i > 0; i >>= 1) v += __shfl_xor(v, i, 32);
  return v;
}
__device__ __forceinline__ unsigned fenc(float f) {
  unsigned u = __float_as_uint(f);
  return (u & 0x80000000u) ? ~u : (u | 0x80000000u);
}
__device__ __forceinline__ float fdec(unsigned u) {
  return (u & 0x80000000u) ? __uint_as_float(u & 0x7FFFFFFFu) : __uint_as_float(~u);
}
__device__ __forceinline__ v16bf load_frag(const bf16* p) {
  union { v16bf v; v8bf h[2]; } u;
  u.h[0] = *reinterpret_cast<const v8bf*>(p);        // K = base .. base+7
  u.h[1] = *reinterpret_cast<const v8bf*>(p + 16);   // K = base+16 .. base+23
  return u.v;
}

// ---------------------------------------------------------------- fills
__global__ void k_fill_f32(float* p, float v, int n) {
  int i = blockIdx.x * blockDim.x + threadIdx.x;
  if (i < n) p[i] = v;
}
__global__ void k_fill_u32(unsigned* p, unsigned v, int n) {
  int i = blockIdx.x * blockDim.x + threadIdx.x;
  if (i < n) p[i] = v;
}

// ---------------------------------------------------------------- weight packing (f32 -> bf16, transposed)
__global__ void k_pack_bt(const float* __restrict__ w, bf16* __restrict__ bt, int K, int N) {
  int idx = blockIdx.x * blockDim.x + threadIdx.x;
  if (idx >= K * N) return;
  int n = idx / K, k = idx - n * K;
  bt[idx] = (bf16)w[(size_t)k * N + n];
}
// combined complex-block matrix: [xr xi] @ [[wr, wi], [-wi, wr]] -> [yr yi], stored transposed
__global__ void k_pack_bt_cplx(const float* __restrict__ wr, const float* __restrict__ wi,
                               bf16* __restrict__ bt, int Kh, int Nh) {
  int K = 2 * Kh, N2 = 2 * Nh;
  int idx = blockIdx.x * blockDim.x + threadIdx.x;
  if (idx >= K * N2) return;
  int n = idx / K, k = idx - n * K;
  float v;
  if (n < Nh) v = (k < Kh) ? wr[(size_t)k * Nh + n] : -wi[(size_t)(k - Kh) * Nh + n];
  else        v = (k < Kh) ? wi[(size_t)k * Nh + (n - Nh)] : wr[(size_t)(k - Kh) * Nh + (n - Nh)];
  bt[idx] = (bf16)v;
}

// ---------------------------------------------------------------- generic WMMA GEMM (fully compile-time shapes)
// A: (M, LDA) bf16; logical K split into k-blocks of width CHAN located at compile-time
// element offsets O0..O5 within each A row. Bt: (N, K) bf16. Wave computes 16 x (16*NT).
// EPI: 0 = store f32, 1 = +bias store f32, 2 = +bias, silu, store bf16
template<int EPI, int K, int CHAN, int NT, int LDA, int LDC,
         int O0 = 0, int O1 = 0, int O2 = 0, int O3 = 0, int O4 = 0, int O5 = 0>
__global__ void __launch_bounds__(128) k_gemm(
    const bf16* __restrict__ A, const bf16* __restrict__ Bt,
    const float* __restrict__ bias,
    float* __restrict__ Cf, bf16* __restrict__ Cb)
{
  constexpr int offs[6] = {O0, O1, O2, O3, O4, O5};
  const int lane = threadIdx.x & 31;
  const int wave = threadIdx.x >> 5;
  const int m0 = (blockIdx.x * 4 + wave) * 16;
  const int n0 = blockIdx.y * (16 * NT);
  const int lr = lane & 15;
  const int lh = lane >> 4;
  v8f acc[NT];
  #pragma unroll
  for (int t = 0; t < NT; ++t) { v8f z = {0.f,0.f,0.f,0.f,0.f,0.f,0.f,0.f}; acc[t] = z; }
  const bf16* arow = A + (size_t)(m0 + lr) * LDA + lh * 8;
  const bf16* brow = Bt + (size_t)(n0 + lr) * K + lh * 8;
  #pragma unroll
  for (int k0 = 0; k0 < K; k0 += 32) {
    const int phys = offs[k0 / CHAN] + (k0 % CHAN);   // compile-time constant after unroll
    v16bf af = load_frag(arow + phys);
    #pragma unroll
    for (int t = 0; t < NT; ++t) {
      v16bf bfrag = load_frag(brow + (size_t)t * 16 * K + k0);
      acc[t] = __builtin_amdgcn_wmma_f32_16x16x32_bf16(
          false, af, false, bfrag, (short)0, acc[t], false, false);
    }
  }
  #pragma unroll
  for (int t = 0; t < NT; ++t) {
    const int n = n0 + t * 16 + lr;
    const float bv = (EPI >= 1) ? bias[n] : 0.f;
    #pragma unroll
    for (int r = 0; r < 8; ++r) {
      const int m = m0 + r + lh * 8;
      float v = acc[t][r] + bv;
      if (EPI == 2) {
        v = v * sigf(v);
        Cb[(size_t)m * LDC + n] = (bf16)v;
      } else {
        Cf[(size_t)m * LDC + n] = v;
      }
    }
  }
}

// ---------------------------------------------------------------- edge feature concat -> bf16
__global__ void k_build_xedge(const float* __restrict__ ed, const float* __restrict__ st,
                              const float* __restrict__ tt, const int* __restrict__ san,
                              const int* __restrict__ tan_, bf16* __restrict__ xe)
{
  int idx = blockIdx.x * blockDim.x + threadIdx.x;
  if (idx >= EDGES * 192) return;
  int e = idx / 192, c = idx - e * 192;
  float v;
  if (c < 64)       v = ed[e * 64 + c];
  else if (c < 128) v = st[san[e] * 64 + (c - 64)];
  else              v = tt[tan_[e] * 64 + (c - 128)];
  xe[idx] = (bf16)v;
}

// ---------------------------------------------------------------- gather + Wigner rotate + radial scale
__global__ void __launch_bounds__(128) k_build_msg(
    const float* __restrict__ x, const int* __restrict__ ei, const float* __restrict__ rad,
    const float* __restrict__ w1, const float* __restrict__ w2, const float* __restrict__ w3,
    bf16* __restrict__ msg)
{
  const int e = blockIdx.x * 4 + (threadIdx.x >> 5);
  const int lane = threadIdx.x & 31;
  const int src = ei[e];
  const int tgt = ei[EDGES + e];
  float m[16][4];
  #pragma unroll
  for (int q = 0; q < 4; ++q) {
    const int c = lane + q * 32;
    const float* xp = (c < 64) ? (x + (size_t)src * 1024 + c) : (x + (size_t)tgt * 1024 + (c - 64));
    #pragma unroll
    for (int j = 0; j < 16; ++j) m[j][q] = xp[j * 64];
  }
  const float* radp = rad + (size_t)e * 512;
  bf16* mp = msg + (size_t)e * 2048;
  #pragma unroll
  for (int q = 0; q < 4; ++q) { int c = lane + q * 32; mp[c] = (bf16)(m[0][q] * radp[c]); }
  const float* W1 = w1 + (size_t)e * 9;
  #pragma unroll
  for (int a = 0; a < 3; ++a) {
    float acc[4] = {0.f,0.f,0.f,0.f};
    #pragma unroll
    for (int b = 0; b < 3; ++b) { float wv = W1[a*3+b];
      #pragma unroll
      for (int q = 0; q < 4; ++q) acc[q] += wv * m[1+b][q]; }
    #pragma unroll
    for (int q = 0; q < 4; ++q) { int c = lane + q*32; mp[(1+a)*128 + c] = (bf16)(acc[q] * radp[128 + c]); }
  }
  const float* W2 = w2 + (size_t)e * 25;
  #pragma unroll
  for (int a = 0; a < 5; ++a) {
    float acc[4] = {0.f,0.f,0.f,0.f};
    #pragma unroll
    for (int b = 0; b < 5; ++b) { float wv = W2[a*5+b];
      #pragma unroll
      for (int q = 0; q < 4; ++q) acc[q] += wv * m[4+b][q]; }
    #pragma unroll
    for (int q = 0; q < 4; ++q) { int c = lane + q*32; mp[(4+a)*128 + c] = (bf16)(acc[q] * radp[256 + c]); }
  }
  const float* W3 = w3 + (size_t)e * 49;
  #pragma unroll
  for (int a = 0; a < 7; ++a) {
    float acc[4] = {0.f,0.f,0.f,0.f};
    #pragma unroll
    for (int b = 0; b < 7; ++b) { float wv = W3[a*7+b];
      #pragma unroll
      for (int q = 0; q < 4; ++q) acc[q] += wv * m[9+b][q]; }
    #pragma unroll
    for (int q = 0; q < 4; ++q) { int c = lane + q*32; mp[(9+a)*128 + c] = (bf16)(acc[q] * radp[384 + c]); }
  }
}

// ---------------------------------------------------------------- attention: LN + leaky-silu + logits + segmax
__global__ void __launch_bounds__(128) k_attn(
    const float* __restrict__ y0, const int* __restrict__ ei,
    const float* __restrict__ lng, const float* __restrict__ lnb, const float* __restrict__ adot,
    float* __restrict__ logits, unsigned* __restrict__ amax)
{
  const int e = blockIdx.x * 4 + (threadIdx.x >> 5);
  const int lane = threadIdx.x & 31;
  const int tgt = ei[EDGES + e];
  const float g = lng[lane], bta = lnb[lane];
  #pragma unroll
  for (int h = 0; h < 8; ++h) {
    float v = y0[(size_t)e * 896 + h * 32 + lane];
    float mu = wsum(v) * (1.f / 32.f);
    float d = v - mu;
    float var = wsum(d * d) * (1.f / 32.f);
    float al = d * rsqrtf(var + 1e-5f) * g + bta;
    float act = 0.2f * al + 0.8f * al * sigf(al);
    float lg = wsum(act * adot[h * 32 + lane]);
    if (lane == 0) {
      logits[e * 8 + h] = lg;
      atomicMax(amax + tgt * 8 + h, fenc(lg));
    }
  }
}

__global__ void k_attn2(const float* __restrict__ logits, const unsigned* __restrict__ amax,
                        const int* __restrict__ ei, float* __restrict__ exb, float* __restrict__ den)
{
  int idx = blockIdx.x * blockDim.x + threadIdx.x;
  if (idx >= EDGES * 8) return;
  int e = idx >> 3, h = idx & 7;
  int tgt = ei[EDGES + e];
  float v = __expf(logits[idx] - fdec(amax[tgt * 8 + h]));
  exb[idx] = v;
  atomicAdd(den + tgt * 8 + h, v);
}

// ---------------------------------------------------------------- scatter so2a outputs back into (E,16,128)
__global__ void k_repack_msg2(const float* __restrict__ y0, const float* __restrict__ ym1,
                              const float* __restrict__ ym2, const float* __restrict__ ym3,
                              float* __restrict__ msg2)
{
  int idx = blockIdx.x * blockDim.x + threadIdx.x;
  if (idx >= EDGES * 16) return;
  int e = idx >> 4, i = idx & 15;
  const float* srcs[4] = {y0, ym1, ym2, ym3};
  int sb = d_sb[i];
  const float4* s4 = (const float4*)(srcs[sb] + (size_t)e * d_strideY[sb] + d_soA[i]);
  float4* dst = (float4*)(msg2 + (size_t)e * 2048 + i * 128);
  #pragma unroll
  for (int q = 0; q < 32; ++q) dst[q] = s4[q];
}

// ---------------------------------------------------------------- SO(3) grid gating + s_out -> bf16 val
__global__ void __launch_bounds__(128) k_grid(
    const float* __restrict__ msg2, const float* __restrict__ y0,
    const float* __restrict__ tg, const float* __restrict__ fg, bf16* __restrict__ valb)
{
  const int e = blockIdx.x * 4 + (threadIdx.x >> 5);
  const int lane = threadIdx.x & 31;
  float gin[16][4];
  const float* mp = msg2 + (size_t)e * 2048;
  #pragma unroll
  for (int n = 0; n < 16; ++n)
    #pragma unroll
    for (int q = 0; q < 4; ++q) gin[n][q] = mp[n * 128 + lane + q * 32];
  float vacc[16][2];
  #pragma unroll
  for (int n = 0; n < 16; ++n) { vacc[n][0] = 0.f; vacc[n][1] = 0.f; }
  #pragma unroll
  for (int p = 0; p < 16; ++p) {
    float gp[4] = {0.f,0.f,0.f,0.f};
    #pragma unroll
    for (int n = 0; n < 16; ++n) { float wv = tg[p * 16 + n];
      #pragma unroll
      for (int q = 0; q < 4; ++q) gp[q] += wv * gin[n][q]; }
    float ga0 = gp[0] * (gp[2] * sigf(gp[2]));
    float ga1 = gp[1] * (gp[3] * sigf(gp[3]));
    #pragma unroll
    for (int n = 0; n < 16; ++n) { float wv = fg[n * 16 + p];
      vacc[n][0] += wv * ga0; vacc[n][1] += wv * ga1; }
  }
  const float* scal = y0 + (size_t)e * 896 + 256;
  #pragma unroll
  for (int q = 0; q < 2; ++q) {
    int c = lane + q * 32;
    vacc[0][q] += scal[c] * (scal[64 + c] * sigf(scal[64 + c]));
  }
  bf16* vp = valb + (size_t)e * 1024;
  #pragma unroll
  for (int n = 0; n < 16; ++n)
    #pragma unroll
    for (int q = 0; q < 2; ++q) vp[n * 64 + lane + q * 32] = (bf16)vacc[n][q];
}

// ---------------------------------------------------------------- attn-weight + inverse rotate + scatter-add
__global__ void __launch_bounds__(128) k_finalize(
    const float* __restrict__ z0, const float* __restrict__ zm1,
    const float* __restrict__ zm2, const float* __restrict__ zm3,
    const float* __restrict__ exb, const float* __restrict__ den, const int* __restrict__ ei,
    const float* __restrict__ w1, const float* __restrict__ w2, const float* __restrict__ w3,
    float* __restrict__ node)
{
  const int e = blockIdx.x * 4 + (threadIdx.x >> 5);
  const int lane = threadIdx.x & 31;
  const int tgt = ei[EDGES + e];
  float at[8];
  #pragma unroll
  for (int h = 0; h < 8; ++h) at[h] = exb[e * 8 + h] / den[tgt * 8 + h];
  float a_q[4];
  #pragma unroll
  for (int q = 0; q < 4; ++q) a_q[q] = (lane < 16) ? at[2 * q] : at[2 * q + 1];
  const float* zs[4] = {z0, zm1, zm2, zm3};
  float v[16][4];
  #pragma unroll
  for (int i = 0; i < 16; ++i) {
    int sb = d_sb[i];
    const float* p = zs[sb] + (size_t)e * d_strideZ[sb] + d_soB[i];
    #pragma unroll
    for (int q = 0; q < 4; ++q) v[i][q] = p[lane + q * 32] * a_q[q];
  }
  float* np = node + (size_t)tgt * 2048;
  #pragma unroll
  for (int q = 0; q < 4; ++q) atomicAdd(np + lane + q * 32, v[0][q]);
  const float* W1 = w1 + (size_t)e * 9;
  #pragma unroll
  for (int a = 0; a < 3; ++a) {
    float acc[4] = {0.f,0.f,0.f,0.f};
    #pragma unroll
    for (int b = 0; b < 3; ++b) { float wv = W1[b * 3 + a];   // W^T
      #pragma unroll
      for (int q = 0; q < 4; ++q) acc[q] += wv * v[1 + b][q]; }
    #pragma unroll
    for (int q = 0; q < 4; ++q) atomicAdd(np + (1 + a) * 128 + lane + q * 32, acc[q]);
  }
  const float* W2 = w2 + (size_t)e * 25;
  #pragma unroll
  for (int a = 0; a < 5; ++a) {
    float acc[4] = {0.f,0.f,0.f,0.f};
    #pragma unroll
    for (int b = 0; b < 5; ++b) { float wv = W2[b * 5 + a];
      #pragma unroll
      for (int q = 0; q < 4; ++q) acc[q] += wv * v[4 + b][q]; }
    #pragma unroll
    for (int q = 0; q < 4; ++q) atomicAdd(np + (4 + a) * 128 + lane + q * 32, acc[q]);
  }
  const float* W3 = w3 + (size_t)e * 49;
  #pragma unroll
  for (int a = 0; a < 7; ++a) {
    float acc[4] = {0.f,0.f,0.f,0.f};
    #pragma unroll
    for (int b = 0; b < 7; ++b) { float wv = W3[b * 7 + a];
      #pragma unroll
      for (int q = 0; q < 4; ++q) acc[q] += wv * v[9 + b][q]; }
    #pragma unroll
    for (int q = 0; q < 4; ++q) atomicAdd(np + (9 + a) * 128 + lane + q * 32, acc[q]);
  }
}

// ---------------------------------------------------------------- final projection per l
__global__ void k_proj(const float* __restrict__ node, const float* __restrict__ pw,
                       const float* __restrict__ pb, float* __restrict__ out)
{
  int idx = blockIdx.x * blockDim.x + threadIdx.x;
  if (idx >= NNODE * 1024) return;
  int o = idx & 63;
  int i = (idx >> 6) & 15;
  int nd = idx >> 10;
  int l = d_lof[i];
  float acc = (i == 0) ? pb[o] : 0.f;
  const float* nrow = node + (size_t)nd * 2048 + i * 128;
  const float* wp = pw + (size_t)l * 8192 + o;
  #pragma unroll 8
  for (int k = 0; k < 128; ++k) acc += nrow[k] * wp[k * 64];
  out[idx] = acc;
}

// ---------------------------------------------------------------- host
extern "C" void kernel_launch(void* const* d_in, const int* in_sizes, int n_in,
                              void* d_out, int out_size, void* d_ws, size_t ws_size,
                              hipStream_t stream)
{
  (void)in_sizes; (void)n_in; (void)out_size; (void)ws_size;
  const float* x    = (const float*)d_in[0];
  const float* ed   = (const float*)d_in[1];
  const float* st   = (const float*)d_in[2];
  const float* tt   = (const float*)d_in[3];
  const float* rw1  = (const float*)d_in[4];
  const float* rb1  = (const float*)d_in[5];
  const float* rw2  = (const float*)d_in[6];
  const float* rb2  = (const float*)d_in[7];
  const float* wg1  = (const float*)d_in[8];
  const float* wg2  = (const float*)d_in[9];
  const float* wg3  = (const float*)d_in[10];
  const float* a_w0 = (const float*)d_in[11];
  const float* a_wr1 = (const float*)d_in[12];
  const float* a_wi1 = (const float*)d_in[13];
  const float* a_wr2 = (const float*)d_in[14];
  const float* a_wi2 = (const float*)d_in[15];
  const float* a_wr3 = (const float*)d_in[16];
  const float* a_wi3 = (const float*)d_in[17];
  const float* lng  = (const float*)d_in[18];
  const float* lnb  = (const float*)d_in[19];
  const float* adot = (const float*)d_in[20];
  const float* tg   = (const float*)d_in[21];
  const float* fg   = (const float*)d_in[22];
  const float* b_w0 = (const float*)d_in[23];
  const float* b_wr1 = (const float*)d_in[24];
  const float* b_wi1 = (const float*)d_in[25];
  const float* b_wr2 = (const float*)d_in[26];
  const float* b_wi2 = (const float*)d_in[27];
  const float* b_wr3 = (const float*)d_in[28];
  const float* b_wi3 = (const float*)d_in[29];
  const float* pw   = (const float*)d_in[30];
  const float* pb   = (const float*)d_in[31];
  const int* san = (const int*)d_in[32];
  const int* tan_ = (const int*)d_in[33];
  const int* ei  = (const int*)d_in[34];
  float* out = (float*)d_out;

  char* w = (char*)d_ws;
  size_t off = 0;
  auto alloc = [&](size_t bytes) -> char* {
    char* p = w + off; off = (off + bytes + 255) & ~(size_t)255; return p;
  };
  bf16* bt_r1 = (bf16*)alloc((size_t)64 * 192 * 2);
  bf16* bt_r2 = (bf16*)alloc((size_t)512 * 64 * 2);
  bf16* bt_a0 = (bf16*)alloc((size_t)896 * 512 * 2);
  bf16* bt_a1 = (bf16*)alloc((size_t)768 * 768 * 2);
  bf16* bt_a2 = (bf16*)alloc((size_t)512 * 512 * 2);
  bf16* bt_a3 = (bf16*)alloc((size_t)256 * 256 * 2);
  bf16* bt_b0 = (bf16*)alloc((size_t)512 * 256 * 2);
  bf16* bt_b1 = (bf16*)alloc((size_t)768 * 384 * 2);
  bf16* bt_b2 = (bf16*)alloc((size_t)512 * 256 * 2);
  bf16* bt_b3 = (bf16*)alloc((size_t)256 * 128 * 2);
  bf16* xedge = (bf16*)alloc((size_t)EDGES * 192 * 2);
  bf16* hbuf  = (bf16*)alloc((size_t)EDGES * 64 * 2);
  float* radp = (float*)alloc((size_t)EDGES * 512 * 4);
  bf16* msgb  = (bf16*)alloc((size_t)EDGES * 2048 * 2);
  float* y0p  = (float*)alloc((size_t)EDGES * 896 * 4);
  float* ym1  = (float*)alloc((size_t)EDGES * 768 * 4);
  float* ym2  = (float*)alloc((size_t)EDGES * 512 * 4);
  float* ym3  = (float*)alloc((size_t)EDGES * 256 * 4);
  float* msg2 = (float*)alloc((size_t)EDGES * 2048 * 4);
  bf16* valb  = (bf16*)alloc((size_t)EDGES * 1024 * 2);
  float* z0p  = (float*)alloc((size_t)EDGES * 512 * 4);
  float* zm1  = (float*)alloc((size_t)EDGES * 768 * 4);
  float* zm2  = (float*)alloc((size_t)EDGES * 512 * 4);
  float* zm3  = (float*)alloc((size_t)EDGES * 256 * 4);
  float* logits = (float*)alloc((size_t)EDGES * 8 * 4);
  float* exb    = (float*)alloc((size_t)EDGES * 8 * 4);
  unsigned* amax = (unsigned*)alloc((size_t)NNODE * 8 * 4);
  float* den    = (float*)alloc((size_t)NNODE * 8 * 4);
  float* nodep  = (float*)alloc((size_t)NNODE * 2048 * 4);

  auto g1 = [](int n) { return dim3((n + 255) / 256); };

  // init
  k_fill_f32<<<g1(NNODE * 2048), 256, 0, stream>>>(nodep, 0.f, NNODE * 2048);
  k_fill_f32<<<g1(NNODE * 8), 256, 0, stream>>>(den, 1e-16f, NNODE * 8);
  k_fill_u32<<<g1(NNODE * 8), 256, 0, stream>>>(amax, 0u, NNODE * 8);

  // pack weights
  k_pack_bt<<<g1(192 * 64), 256, 0, stream>>>(rw1, bt_r1, 192, 64);
  k_pack_bt<<<g1(64 * 512), 256, 0, stream>>>(rw2, bt_r2, 64, 512);
  k_pack_bt<<<g1(512 * 896), 256, 0, stream>>>(a_w0, bt_a0, 512, 896);
  k_pack_bt<<<g1(256 * 512), 256, 0, stream>>>(b_w0, bt_b0, 256, 512);
  k_pack_bt_cplx<<<g1(768 * 768), 256, 0, stream>>>(a_wr1, a_wi1, bt_a1, 384, 384);
  k_pack_bt_cplx<<<g1(512 * 512), 256, 0, stream>>>(a_wr2, a_wi2, bt_a2, 256, 256);
  k_pack_bt_cplx<<<g1(256 * 256), 256, 0, stream>>>(a_wr3, a_wi3, bt_a3, 128, 128);
  k_pack_bt_cplx<<<g1(768 * 384), 256, 0, stream>>>(b_wr1, b_wi1, bt_b1, 192, 384);
  k_pack_bt_cplx<<<g1(512 * 256), 256, 0, stream>>>(b_wr2, b_wi2, bt_b2, 128, 256);
  k_pack_bt_cplx<<<g1(256 * 128), 256, 0, stream>>>(b_wr3, b_wi3, bt_b3, 64, 128);

  // radial MLP
  k_build_xedge<<<g1(EDGES * 192), 256, 0, stream>>>(ed, st, tt, san, tan_, xedge);
  k_gemm<2, 192, 192, 4, 192, 64><<<dim3(EDGES / 64, 1), 128, 0, stream>>>(xedge, bt_r1, rb1, nullptr, hbuf);
  k_gemm<1, 64, 64, 8, 64, 512><<<dim3(EDGES / 64, 4), 128, 0, stream>>>(hbuf, bt_r2, rb2, radp, nullptr);

  // rotate + scale
  k_build_msg<<<EDGES / 4, 128, 0, stream>>>(x, ei, radp, wg1, wg2, wg3, msgb);

  // so2a (WMMA)
  k_gemm<0, 512, 128, 8, 2048, 896, 0, 256, 768, 1536>
      <<<dim3(EDGES / 64, 7), 128, 0, stream>>>(msgb, bt_a0, nullptr, y0p, nullptr);
  k_gemm<0, 768, 128, 8, 2048, 768, 384, 896, 1664, 128, 640, 1408>
      <<<dim3(EDGES / 64, 6), 128, 0, stream>>>(msgb, bt_a1, nullptr, ym1, nullptr);
  k_gemm<0, 512, 128, 8, 2048, 512, 1024, 1792, 512, 1280>
      <<<dim3(EDGES / 64, 4), 128, 0, stream>>>(msgb, bt_a2, nullptr, ym2, nullptr);
  k_gemm<0, 256, 128, 8, 2048, 256, 1920, 1152>
      <<<dim3(EDGES / 64, 2), 128, 0, stream>>>(msgb, bt_a3, nullptr, ym3, nullptr);

  // attention softmax over segments
  k_attn<<<EDGES / 4, 128, 0, stream>>>(y0p, ei, lng, lnb, adot, logits, amax);
  k_attn2<<<g1(EDGES * 8), 256, 0, stream>>>(logits, amax, ei, exb, den);

  // grid nonlinearity
  k_repack_msg2<<<g1(EDGES * 16), 256, 0, stream>>>(y0p, ym1, ym2, ym3, msg2);
  k_grid<<<EDGES / 4, 128, 0, stream>>>(msg2, y0p, tg, fg, valb);

  // so2b (WMMA)
  k_gemm<0, 256, 64, 8, 1024, 512, 0, 128, 384, 768>
      <<<dim3(EDGES / 64, 4), 128, 0, stream>>>(valb, bt_b0, nullptr, z0p, nullptr);
  k_gemm<0, 384, 64, 8, 1024, 768, 192, 448, 832, 64, 320, 704>
      <<<dim3(EDGES / 64, 6), 128, 0, stream>>>(valb, bt_b1, nullptr, zm1, nullptr);
  k_gemm<0, 256, 64, 8, 1024, 512, 512, 896, 256, 640>
      <<<dim3(EDGES / 64, 4), 128, 0, stream>>>(valb, bt_b2, nullptr, zm2, nullptr);
  k_gemm<0, 128, 64, 8, 1024, 256, 960, 576>
      <<<dim3(EDGES / 64, 2), 128, 0, stream>>>(valb, bt_b3, nullptr, zm3, nullptr);

  // weight, inverse rotate, scatter, project
  k_finalize<<<EDGES / 4, 128, 0, stream>>>(z0p, zm1, zm2, zm3, exb, den, ei, wg1, wg2, wg3, nodep);
  k_proj<<<g1(NNODE * 1024), 256, 0, stream>>>(nodep, pw, pb, out);
}
